// GraphMoEPriorOnly_10101763080591
// MI455X (gfx1250) — matrix operand order repaired and checked by reading, not compiled, not run on hardware
//
#include <hip/hip_runtime.h>
#include <hip/hip_bf16.h>
#include <math.h>

// ---------------------------------------------------------------------------
// GraphMoE (prior-only routing) for MI455X / gfx1250, wave32 + WMMA bf16.
//   h = relu(x @ W_enc + b_enc)
//   probs = softmax(-(logn - centers)^2)            [N,4]
//   per expert e (layers l=0..2):
//     agg = scatter_add(h[src] -> dst)              (layer-0 agg shared by all experts)
//     h   = act(h @ Ws[e,l] + agg @ Wn[e,l] + b[e,l])
//   out += probs[:,e] * h_final_e
// ---------------------------------------------------------------------------

typedef __attribute__((ext_vector_type(16))) __bf16 v16bf;
typedef __attribute__((ext_vector_type(8)))  __bf16 v8bf;
typedef __attribute__((ext_vector_type(8)))  float  v8f;

#define HID 128
#define NEXP 4
#define NLAYERS 3
#define INV_LOGN_SCALE 0.10857362047581294f  // 1/log(1e4)

// ------------------------- graph-size prior -------------------------------
__global__ void count_kernel(const int* __restrict__ batch, int* __restrict__ counts, int N) {
    int n = blockIdx.x * blockDim.x + threadIdx.x;
    if (n < N) atomicAdd(&counts[batch[n]], 1);
}

__global__ void probs_kernel(const int* __restrict__ batch, const int* __restrict__ counts,
                             const float* __restrict__ centers, float* __restrict__ probs, int N) {
    int n = blockIdx.x * blockDim.x + threadIdx.x;
    if (n >= N) return;
    float c = (float)counts[batch[n]];
    float logn = logf(fmaxf(c, 1.0f)) * INV_LOGN_SCALE;
    float z[NEXP];
    float m = -1e30f;
    #pragma unroll
    for (int e = 0; e < NEXP; ++e) {
        float d = logn - centers[e];
        z[e] = -d * d;
        m = fmaxf(m, z[e]);
    }
    float s = 0.0f;
    #pragma unroll
    for (int e = 0; e < NEXP; ++e) { z[e] = __expf(z[e] - m); s += z[e]; }
    float inv = 1.0f / s;
    #pragma unroll
    for (int e = 0; e < NEXP; ++e) probs[n * NEXP + e] = z[e] * inv;
}

// ------------------------------ encoder -----------------------------------
__global__ void encoder_kernel(const float* __restrict__ x, const float* __restrict__ W,
                               const float* __restrict__ b, float* __restrict__ h, int N) {
    int idx = blockIdx.x * blockDim.x + threadIdx.x;
    if (idx >= N * HID) return;
    int n = idx >> 7, j = idx & 127;
    const float* xr = x + (size_t)n * 6;
    float acc = b[j];
    #pragma unroll
    for (int k = 0; k < 6; ++k) acc += xr[k] * W[k * HID + j];
    h[idx] = fmaxf(acc, 0.0f);
}

// --------------------- edge gather + scatter-add --------------------------
// one wave32 per edge: lane reads float4 of h[src], atomically adds into agg[dst]
__global__ void aggregate_kernel(const float* __restrict__ h, const int* __restrict__ src,
                                 const int* __restrict__ dst, float* __restrict__ agg, int E) {
    int t = blockIdx.x * blockDim.x + threadIdx.x;
    int e = t >> 5;
    if (e >= E) return;
    int lane = t & 31;
    int s = src[e], d = dst[e];
    float4 v = ((const float4*)(h + (size_t)s * HID))[lane];
    float* a = agg + (size_t)d * HID + lane * 4;
    atomicAdd(a + 0, v.x);
    atomicAdd(a + 1, v.y);
    atomicAdd(a + 2, v.z);
    atomicAdd(a + 3, v.w);
}

// -------------------- dual-GEMM layer via WMMA bf16 -----------------------
// Y[N,128] = act( H @ Ws + AGG @ Wn + bias ), concatenated K = 256.
// 128 threads = 4 waves; each wave: 16 rows x 128 cols. Weights in 64KB LDS
// as bf16, K-chunked so B fragments are contiguous 32B runs.
__global__ void __launch_bounds__(128)
gemm_dual_wmma_kernel(const float* __restrict__ H, const float* __restrict__ AGG,
                      const float* __restrict__ Ws, const float* __restrict__ Wn,
                      const float* __restrict__ bias, float* __restrict__ Y,
                      int N, int applyRelu) {
    // layout: idx = (k/16)*2048 + n*16 + (k%16), combined k in [0,256): Ws then Wn
    __shared__ __bf16 Wlds[32768];  // 64 KB exactly

    const int tid = threadIdx.x;
    for (int kn = tid; kn < HID * HID; kn += 128) {
        int n = kn & 127, k = kn >> 7;
        Wlds[((k >> 4) << 11) + (n << 4) + (k & 15)] = (__bf16)Ws[kn];
        int k2 = k + HID;
        Wlds[((k2 >> 4) << 11) + (n << 4) + (k2 & 15)] = (__bf16)Wn[kn];
    }
    __syncthreads();

    const int wave = tid >> 5;
    const int lane = tid & 31;
    const int half = lane >> 4;   // lane group: 0 -> lanes 0-15, 1 -> lanes 16-31
    const int lr   = lane & 15;
    const int rowbase = blockIdx.x * 64 + wave * 16;

    int arow = rowbase + lr;                 // A-matrix row for this lane
    if (arow > N - 1) arow = N - 1;          // clamp (stores are guarded)

    v8f acc[8];
    const v8f vz = {0.f, 0.f, 0.f, 0.f, 0.f, 0.f, 0.f, 0.f};
    #pragma unroll
    for (int t = 0; t < 8; ++t) acc[t] = vz;

    #pragma unroll
    for (int kc = 0; kc < 256; kc += 32) {   // concatenated K
        const float* srcp = (kc < HID) ? H : AGG;
        const int kk = kc & 127;
        // A fragment, contiguous K order (same as B): lane half h holds K = 16h..16h+15
        const float* rp = srcp + (size_t)arow * HID + kk + half * 16;
        float4 a0 = *(const float4*)(rp);
        float4 a1 = *(const float4*)(rp + 4);
        float4 a2 = *(const float4*)(rp + 8);
        float4 a3 = *(const float4*)(rp + 12);
        v16bf A;
        A[0]  = (__bf16)a0.x; A[1]  = (__bf16)a0.y; A[2]  = (__bf16)a0.z; A[3]  = (__bf16)a0.w;
        A[4]  = (__bf16)a1.x; A[5]  = (__bf16)a1.y; A[6]  = (__bf16)a1.z; A[7]  = (__bf16)a1.w;
        A[8]  = (__bf16)a2.x; A[9]  = (__bf16)a2.y; A[10] = (__bf16)a2.z; A[11] = (__bf16)a2.w;
        A[12] = (__bf16)a3.x; A[13] = (__bf16)a3.y; A[14] = (__bf16)a3.z; A[15] = (__bf16)a3.w;

        const int kchunk = (kc >> 4) + half;  // lane half h covers K = kc+16h .. kc+16h+15
        #pragma unroll
        for (int nt = 0; nt < 8; ++nt) {
            int n = (nt << 4) + lr;           // B column for this lane
            const v8bf* bp = (const v8bf*)&Wlds[(kchunk << 11) + (n << 4)];
            v8bf blo = bp[0];
            v8bf bhi = bp[1];
            v16bf B;
            #pragma unroll
            for (int i = 0; i < 8; ++i) { B[i] = blo[i]; B[8 + i] = bhi[i]; }
            acc[nt] = __builtin_amdgcn_wmma_f32_16x16x32_bf16(
                false, A, false, B, (short)0, acc[nt], false, false);
        }
    }

    // epilogue: C/D f32 layout — lanes 0-15: M = r; lanes 16-31: M = 8 + r; N = lr
    #pragma unroll
    for (int nt = 0; nt < 8; ++nt) {
        int col = (nt << 4) + lr;
        float bv = bias[col];
        #pragma unroll
        for (int r = 0; r < 8; ++r) {
            int row = rowbase + half * 8 + r;
            if (row < N) {
                float v = acc[nt][r] + bv;
                if (applyRelu) v = fmaxf(v, 0.0f);
                Y[(size_t)row * HID + col] = v;
            }
        }
    }
}

// ------------------------ MoE mix (accumulate) ----------------------------
__global__ void mix_kernel(const float* __restrict__ y, const float* __restrict__ probs,
                           float* __restrict__ out, int N, int e) {
    int idx = blockIdx.x * blockDim.x + threadIdx.x;
    if (idx >= N * HID) return;
    int n = idx >> 7;
    out[idx] += probs[n * NEXP + e] * y[idx];
}

// ---------------------------------------------------------------------------
extern "C" void kernel_launch(void* const* d_in, const int* in_sizes, int n_in,
                              void* d_out, int out_size, void* d_ws, size_t ws_size,
                              hipStream_t stream) {
    const float* x       = (const float*)d_in[0];
    const int*   ei      = (const int*)d_in[1];   // [2,E]: src row then dst row
    const int*   batch   = (const int*)d_in[2];
    const float* W_enc   = (const float*)d_in[3];
    const float* b_enc   = (const float*)d_in[4];
    const float* W_self  = (const float*)d_in[5]; // [4,3,128,128]
    const float* W_nbr   = (const float*)d_in[6]; // [4,3,128,128]
    const float* b_conv  = (const float*)d_in[7]; // [4,3,128]
    const float* centers = (const float*)d_in[8]; // [4]

    const int N = in_sizes[2];
    const int E = in_sizes[1] / 2;
    const int* src = ei;
    const int* dst = ei + E;

    char*  ws = (char*)d_ws;
    size_t sz = (size_t)N * HID * sizeof(float);
    float* h0     = (float*)(ws + 0 * sz);
    float* agg0   = (float*)(ws + 1 * sz);   // layer-0 aggregation, shared by all experts
    float* hA     = (float*)(ws + 2 * sz);
    float* hB     = (float*)(ws + 3 * sz);
    float* aggL   = (float*)(ws + 4 * sz);
    float* probs  = (float*)(ws + 5 * sz);
    int*   counts = (int*)  (ws + 5 * sz + (size_t)N * NEXP * sizeof(float));

    const int nbN  = (N + 255) / 256;
    const int nel  = N * HID;
    const int nbEl = (nel + 255) / 256;
    const int nbAg = (int)(((size_t)E * 32 + 255) / 256);
    const int nbGm = (N + 63) / 64;

    // graph-size prior
    hipMemsetAsync(counts, 0, 16 * sizeof(int), stream);
    count_kernel<<<nbN, 256, 0, stream>>>(batch, counts, N);
    probs_kernel<<<nbN, 256, 0, stream>>>(batch, counts, centers, probs, N);

    // encoder + shared layer-0 aggregation
    encoder_kernel<<<nbEl, 256, 0, stream>>>(x, W_enc, b_enc, h0, N);
    hipMemsetAsync(agg0, 0, sz, stream);
    aggregate_kernel<<<nbAg, 256, 0, stream>>>(h0, src, dst, agg0, E);

    // output accumulator
    hipMemsetAsync(d_out, 0, (size_t)out_size * sizeof(float), stream);

    for (int e = 0; e < NEXP; ++e) {
        const float* Ws0 = W_self + ((size_t)e * NLAYERS + 0) * HID * HID;
        const float* Ws1 = W_self + ((size_t)e * NLAYERS + 1) * HID * HID;
        const float* Ws2 = W_self + ((size_t)e * NLAYERS + 2) * HID * HID;
        const float* Wn0 = W_nbr  + ((size_t)e * NLAYERS + 0) * HID * HID;
        const float* Wn1 = W_nbr  + ((size_t)e * NLAYERS + 1) * HID * HID;
        const float* Wn2 = W_nbr  + ((size_t)e * NLAYERS + 2) * HID * HID;
        const float* bc0 = b_conv + ((size_t)e * NLAYERS + 0) * HID;
        const float* bc1 = b_conv + ((size_t)e * NLAYERS + 1) * HID;
        const float* bc2 = b_conv + ((size_t)e * NLAYERS + 2) * HID;

        // layer 0 (shared agg0)
        gemm_dual_wmma_kernel<<<nbGm, 128, 0, stream>>>(h0, agg0, Ws0, Wn0, bc0, hA, N, 1);
        // layer 1
        hipMemsetAsync(aggL, 0, sz, stream);
        aggregate_kernel<<<nbAg, 256, 0, stream>>>(hA, src, dst, aggL, E);
        gemm_dual_wmma_kernel<<<nbGm, 128, 0, stream>>>(hA, aggL, Ws1, Wn1, bc1, hB, N, 1);
        // layer 2 (no relu)
        hipMemsetAsync(aggL, 0, sz, stream);
        aggregate_kernel<<<nbAg, 256, 0, stream>>>(hB, src, dst, aggL, E);
        gemm_dual_wmma_kernel<<<nbGm, 128, 0, stream>>>(hB, aggL, Ws2, Wn2, bc2, hA, N, 0);
        // MoE blend
        mix_kernel<<<nbEl, 256, 0, stream>>>(hA, probs, (float*)d_out, N, e);
    }
}
